// ManyNetworkModel_71717363908882
// MI455X (gfx1250) — matrix operand order repaired and checked by reading, not compiled, not run on hardware
//
#include <hip/hip_runtime.h>
#include <hip/hip_bf16.h>
#include <stdint.h>

// Problem constants (from reference)
#define BATCH   4096
#define NFEAT   128     // M: number of per-feature models == out cols
#define NACT    16
#define HIDDEN  256
#define IN_DIM  144     // NFEAT + NACT
#define K_PAD   160     // IN_DIM padded to multiple of 32 for 16x16x32 WMMA
#define BT      128     // batch rows per block

typedef __attribute__((ext_vector_type(16))) __bf16         v16bf;
typedef __attribute__((ext_vector_type(8)))  float          v8f;
typedef __attribute__((ext_vector_type(8)))  unsigned short u16x8;
typedef __attribute__((ext_vector_type(4)))  unsigned int   u32x4;
typedef __attribute__((ext_vector_type(8)))  int            i32x8;
typedef __attribute__((ext_vector_type(4)))  int            i32x4;

union Frag {
    u16x8 h[2];
    v16bf v;
};

#if defined(__has_builtin)
#if __has_builtin(__builtin_amdgcn_tensor_load_to_lds) && \
    __has_builtin(__builtin_amdgcn_s_wait_tensorcnt) && \
    __has_builtin(__builtin_amdgcn_groupstaticsize)
#define HAVE_TDM 1
#endif
#endif

static __device__ __forceinline__ unsigned short f2bf(float f) {
    // round-to-nearest-even f32 -> bf16
    unsigned int u = __builtin_bit_cast(unsigned int, f);
    u += 0x7FFFu + ((u >> 16) & 1u);
    return (unsigned short)(u >> 16);
}

#ifdef HAVE_TDM
// Issue a TDM DMA of a contiguous region (n_qwords * 8 bytes) from global
// memory into LDS at byte offset lds_off. Descriptor per CDNA5 ISA §8:
// 2D tensor, one row of n_qwords 8-byte elements.
static __device__ __forceinline__ void tdm_load_contig(unsigned lds_off,
                                                       const void* gptr,
                                                       unsigned n_qwords) {
    unsigned long long ga = (unsigned long long)gptr;
    u32x4 g0;
    g0[0] = 1u;                                   // count=1, user mode
    g0[1] = lds_off;                              // lds_addr [63:32]
    g0[2] = (unsigned)(ga & 0xFFFFFFFFu);         // global_addr[31:0]
    g0[3] = (unsigned)((ga >> 32) & 0x01FFFFFFu)  // global_addr[56:32]
            | (2u << 30);                         // type=2 ("image")
    i32x8 g1;
    g1[0] = (3 << 16);                            // data_size=3 (8B); mask=0
    g1[1] = (int)((n_qwords & 0xFFFFu) << 16);    // tensor_dim0[15:0] @bit48
    g1[2] = (int)((n_qwords >> 16) & 0xFFFFu)     // tensor_dim0[31:16]
            | (1 << 16);                          // tensor_dim1 = 1 @bit80
    g1[3] = (int)((n_qwords & 0xFFFFu) << 16);    // tile_dim0 @bit112
    g1[4] = 1;                                    // tile_dim1 = 1 @bit128
    g1[5] = (int)n_qwords;                        // tensor_dim0_stride[31:0]
    g1[6] = 0;                                    // stride hi / dim1_stride lo
    g1[7] = 0;
    i32x4 z4 = {0, 0, 0, 0};
#if __clang_major__ >= 23
    i32x8 z8 = {0, 0, 0, 0, 0, 0, 0, 0};
    __builtin_amdgcn_tensor_load_to_lds(g0, g1, z4, z4, z8, 0);
#else
    __builtin_amdgcn_tensor_load_to_lds(g0, g1, z4, z4, 0);
#endif
}
#endif

// ---------------------------------------------------------------------------
// Prep 1: fa = concat(f_t, a_t) -> bf16, K padded 144 -> 160 with zeros.
// ---------------------------------------------------------------------------
__global__ __launch_bounds__(256) void prep_fa_kernel(
    const float* __restrict__ f_t,   // [4096][128]
    const float* __restrict__ a_t,   // [4096][16]
    unsigned short* __restrict__ faB)
{
    int idx = blockIdx.x * 256 + threadIdx.x;
    if (idx >= BATCH * K_PAD) return;
    int b = idx / K_PAD;
    int k = idx - b * K_PAD;
    float v = 0.f;
    if (k < NFEAT)        v = f_t[b * NFEAT + k];
    else if (k < IN_DIM)  v = a_t[b * NACT + (k - NFEAT)];
    faB[idx] = f2bf(v);
}

// ---------------------------------------------------------------------------
// Prep 2: W1 [m][k=144][h=256] f32 -> bf16 transposed+padded: w1B[m][h=256][k=160]
// ---------------------------------------------------------------------------
__global__ __launch_bounds__(256) void prep_w1_kernel(
    const float* __restrict__ W1,    // [128][144][256]
    unsigned short* __restrict__ w1B)
{
    int idx = blockIdx.x * 256 + threadIdx.x;
    if (idx >= NFEAT * HIDDEN * K_PAD) return;
    int m = idx / (HIDDEN * K_PAD);
    int r = idx - m * (HIDDEN * K_PAD);
    int h = r / K_PAD;
    int k = r - h * K_PAD;
    unsigned short v = 0;
    if (k < IN_DIM) v = f2bf(W1[(m * IN_DIM + k) * HIDDEN + h]);
    w1B[idx] = v;
}

// ---------------------------------------------------------------------------
// Main fused kernel: per block = (one model m) x (128 batch rows).
//   h_tile = relu(fa_tile @ W1[m] + b1[m])   (WMMA bf16 -> f32 accumulators)
//   out    = h_tile @ W2[m] + b2[m] + f_t    (per-lane FMA + shfl reduce)
// LDS: sA 40960B | sB 81920B | sW2 1024B | sB1 1024B = 124928B dynamic.
// ---------------------------------------------------------------------------
__global__ __launch_bounds__(256) void mlp_fused_kernel(
    const unsigned short* __restrict__ faB,  // [4096][160] bf16
    const unsigned short* __restrict__ w1B,  // [128][256][160] bf16
    const float* __restrict__ b1,            // [128][256]
    const float* __restrict__ w2,            // [128][256]
    const float* __restrict__ b2,            // [128]
    const float* __restrict__ f_t,           // [4096][128]
    float* __restrict__ out)                 // [4096][128]
{
    extern __shared__ char smem[];
    unsigned short* sA  = (unsigned short*)(smem);            // 128 x 160 bf16
    unsigned short* sB  = (unsigned short*)(smem + 40960);    // 256 x 160 bf16
    float*          sW2 = (float*)(smem + 122880);            // 256 f32
    float*          sB1 = (float*)(smem + 123904);            // 256 f32

    const int bt  = blockIdx.x;
    const int m   = blockIdx.y;
    const int b0  = bt * BT;
    const int tid = threadIdx.x;

#ifdef HAVE_TDM
    // ---- Stage fa tile (40960 B) + W1^T[m] (81920 B) via Tensor Data Mover.
    // Wave 0 issues both DMAs (TDM ignores EXEC; scalar-uniform descriptors),
    // drains TENSORcnt, then everyone syncs at the workgroup barrier.
    if (tid < 32) {
        const unsigned lds_base = __builtin_amdgcn_groupstaticsize();
        tdm_load_contig(lds_base,         faB + (size_t)b0 * K_PAD,        40960u / 8u);
        tdm_load_contig(lds_base + 40960, w1B + (size_t)m * (HIDDEN * K_PAD), 81920u / 8u);
        __builtin_amdgcn_s_wait_tensorcnt(0);
    }
#else
    // ---- Fallback: vector-copy staging through VGPRs ----
    {
        const uint4* src = (const uint4*)(faB + (size_t)b0 * K_PAD);
        uint4*       dst = (uint4*)sA;
        #pragma unroll
        for (int i = 0; i < 10; ++i) dst[tid + 256 * i] = src[tid + 256 * i];
    }
    {
        const uint4* src = (const uint4*)(w1B + (size_t)m * (HIDDEN * K_PAD));
        uint4*       dst = (uint4*)sB;
        #pragma unroll
        for (int i = 0; i < 20; ++i) dst[tid + 256 * i] = src[tid + 256 * i];
    }
#endif
    // ---- Stage W2[m], b1[m] (small, regular path) ----
    if (tid < 64) {
        ((uint4*)sW2)[tid] = ((const uint4*)(w2 + (size_t)m * HIDDEN))[tid];
        ((uint4*)sB1)[tid] = ((const uint4*)(b1 + (size_t)m * HIDDEN))[tid];
    }
    __syncthreads();

    const int lane = tid & 31;
    const int wave = tid >> 5;     // 0..7
    const int half = lane >> 4;    // 0|1
    const int l16  = lane & 15;

    // ---- Preload A fragments for all 5 K-tiles (reused across 16 N-tiles).
    // A 16-bit 16x32 layout (ISA 7.12.2): lane holds row M=lane%16; K segs
    // [kb,kb+8) and [kb+16,kb+24) with kb = 8*(lane/16).
    Frag afrag[5];
    {
        const int arow = wave * 16 + l16;
        #pragma unroll
        for (int kt = 0; kt < 5; ++kt) {
            const unsigned short* p = sA + arow * K_PAD + kt * 32 + half * 8;
            afrag[kt].h[0] = *(const u16x8*)(p);
            afrag[kt].h[1] = *(const u16x8*)(p + 16);
        }
    }

    // ---- Double-buffered B fragments: load tile nt+1 while tile nt's WMMAs
    // execute, so the pre-WMMA s_wait_dscnt tolerates 10 in-flight DS loads.
    Frag bf[2][5];
    const unsigned short* sBlane = sB + l16 * K_PAD + half * 8;  // col l16
    {
        #pragma unroll
        for (int kt = 0; kt < 5; ++kt) {
            const unsigned short* p = sBlane + kt * 32;           // nt = 0
            bf[0][kt].h[0] = *(const u16x8*)(p);
            bf[0][kt].h[1] = *(const u16x8*)(p + 16);
        }
    }

    float acc2[8];
    #pragma unroll
    for (int i = 0; i < 8; ++i) acc2[i] = 0.f;

    #pragma unroll 4
    for (int nt = 0; nt < 16; ++nt) {
        const int cur = nt & 1;
        if (nt + 1 < 16) {
            const unsigned short* nb = sBlane + (nt + 1) * 16 * K_PAD;
            #pragma unroll
            for (int kt = 0; kt < 5; ++kt) {
                const unsigned short* p = nb + kt * 32;
                bf[cur ^ 1][kt].h[0] = *(const u16x8*)(p);
                bf[cur ^ 1][kt].h[1] = *(const u16x8*)(p + 16);
            }
        }
        v8f c = {0.f, 0.f, 0.f, 0.f, 0.f, 0.f, 0.f, 0.f};
        #pragma unroll
        for (int kt = 0; kt < 5; ++kt) {
            c = __builtin_amdgcn_wmma_f32_16x16x32_bf16(
                    false, afrag[kt].v, false, bf[cur][kt].v,
                    (short)0, c, false, false);
        }
        // Fused second einsum: bias + relu + scale by W2[n], per-lane partial.
        const int   n  = nt * 16 + l16;
        const float wv = sW2[n];
        const float bv = sB1[n];
        #pragma unroll
        for (int i = 0; i < 8; ++i) {
            float h = c[i] + bv;          // C/D layout: VGPR i = row i+8*half
            h = h > 0.f ? h : 0.f;        // relu
            acc2[i] += h * wv;            // partial dot over this lane's cols
        }
    }

    // ---- Reduce the 16 per-column partials across lanes of each half-wave,
    // then skip-connection + biases + store. ----
    const float b2v = b2[m];
    #pragma unroll
    for (int i = 0; i < 8; ++i) {
        float v = acc2[i];
        v += __shfl_xor(v, 1);
        v += __shfl_xor(v, 2);
        v += __shfl_xor(v, 4);
        v += __shfl_xor(v, 8);   // masks 1..8 stay within each 16-lane half
        if (l16 == 0) {
            const int row = b0 + wave * 16 + half * 8 + i;
            out[row * NFEAT + m] = v + b2v + f_t[row * NFEAT + m];
        }
    }
}

// ---------------------------------------------------------------------------
extern "C" void kernel_launch(void* const* d_in, const int* in_sizes, int n_in,
                              void* d_out, int out_size, void* d_ws, size_t ws_size,
                              hipStream_t stream) {
    const float* f_t = (const float*)d_in[0];
    const float* a_t = (const float*)d_in[1];
    const float* W1  = (const float*)d_in[2];
    const float* b1  = (const float*)d_in[3];
    const float* W2  = (const float*)d_in[4];
    const float* b2  = (const float*)d_in[5];
    float* out = (float*)d_out;

    // Workspace layout (bf16 staging):
    //   faB : BATCH * K_PAD          = 655,360 elems (1,310,720 B)
    //   w1B : NFEAT * HIDDEN * K_PAD = 5,242,880 elems (10,485,760 B)
    unsigned short* faB = (unsigned short*)d_ws;
    unsigned short* w1B = faB + (size_t)BATCH * K_PAD;

    {
        int total = BATCH * K_PAD;
        prep_fa_kernel<<<(total + 255) / 256, 256, 0, stream>>>(f_t, a_t, faB);
    }
    {
        int total = NFEAT * HIDDEN * K_PAD;
        prep_w1_kernel<<<(total + 255) / 256, 256, 0, stream>>>(W1, w1B);
    }
    {
        dim3 grid(BATCH / BT, NFEAT);  // 32 x 128 blocks
        size_t lds = 124928;           // sA + sB + sW2 + sB1
        mlp_fused_kernel<<<grid, 256, lds, stream>>>(faB, w1B, b1, W2, b2, f_t, out);
    }
}
